// NearestTokenLookUp_31147102831265
// MI455X (gfx1250) — matrix operand order repaired and checked by reading, not compiled, not run on hardware
//
#include <hip/hip_runtime.h>

// ---------------------------------------------------------------------------
// NearestTokenLookUp for MI455X (gfx1250, wave32, WMMA + TDM / async-to-LDS).
// score(q, v) = ||e_v||^2 - 2 q.e_v ; argmin over v; gather e_v.
// GEMM via v_wmma_f32_16x16x32_bf16 with bf16 hi/lo split (3 products) for
// ~fp32 accuracy; vocab tiles streamed to double-buffered LDS either by the
// Tensor Data Mover (TENSOR_LOAD_TO_LDS, TENSORcnt) or, if the builtin is
// unavailable, by per-lane GLOBAL_LOAD_ASYNC_TO_LDS_B128 (ASYNCcnt).
// ---------------------------------------------------------------------------

typedef __bf16 bf16;
typedef __attribute__((ext_vector_type(8)))  bf16  v8bf;
typedef __attribute__((ext_vector_type(16))) bf16  v16bf;
typedef __attribute__((ext_vector_type(8)))  float v8f;

union BF16x16 { v16bf v; v8bf h[2]; };

#define DEVFN __device__ __forceinline__

// problem constants
#define BB   2
#define LL   2048
#define DD   512
#define VV   50257
#define NQ   (BB * LL)                 // 4096 queries
#define NCHUNK 16                      // vocab chunks (grid.y of main kernel)
#define VT   32                        // vocab rows per tile iteration
#define TILES_PER_CHUNK 101
#define ROWS_PER_CHUNK (VT * TILES_PER_CHUNK)   // 3232
#define VPAD (NCHUNK * ROWS_PER_CHUNK)          // 51712 >= VV

// LDS layout (identical for TDM and async paths, and for host/device passes):
// per plane: 32 rows x (1024 data + 16 pad) bytes  -> bank-friendly stride.
#define LDS_ROW_STRIDE 1040
#define PLANE_BYTES (VT * LDS_ROW_STRIDE)        // 33280
#define LBUF_BYTES  (2 * PLANE_BYTES)            // hi + lo planes = 66560
#define SMEM_BYTES  (2 * LBUF_BYTES)             // double buffer = 133120

// TDM availability (device pass only; layout does not depend on this)
#if defined(__has_builtin)
#  if __has_builtin(__builtin_amdgcn_tensor_load_to_lds)
#    define USE_TDM 1
#  else
#    define USE_TDM 0
#  endif
#else
#  define USE_TDM 0
#endif

// fp32 -> bf16 round-to-nearest-even (no reliance on __bf16 arithmetic)
DEVFN bf16 f2bf(float x) {
  unsigned u = __float_as_uint(x);
  unsigned r = (u + 0x7FFFu + ((u >> 16) & 1u)) >> 16;
  unsigned short s = (unsigned short)r;
  return __builtin_bit_cast(bf16, s);
}
DEVFN float bf2f(bf16 h) {
  unsigned short s = __builtin_bit_cast(unsigned short, h);
  return __uint_as_float(((unsigned)s) << 16);
}

// ---- fallback: one 16B async DMA global -> LDS (ASYNCcnt-tracked) ---------
DEVFN void async_cp16(unsigned lds_addr, const void* gsrc) {
  const unsigned long long ga = (unsigned long long)gsrc;
  asm volatile("global_load_async_to_lds_b128 %0, %1, off"
               :: "v"(lds_addr), "v"(ga) : "memory");
}
DEVFN void wait_async_le32() { asm volatile("s_wait_asynccnt 0x20" ::: "memory"); }
DEVFN void wait_async_0()    { asm volatile("s_wait_asynccnt 0x0"  ::: "memory"); }

#if USE_TDM
typedef unsigned int v4u __attribute__((ext_vector_type(4)));
typedef int          v8i __attribute__((ext_vector_type(8)));
typedef int          v4i __attribute__((ext_vector_type(4)));

// TDM: load one 32-row x 1024B plane into LDS with 16B padding per row
// (pad_interval code 7 = 256 DWORDs, pad_amount code 3 = 4 DWORDs) so the
// LDS row stride becomes 1040B. D# packed per CDNA5 ISA ch.8 (2-D tile).
DEVFN void tdm_load_plane(const void* gbase, unsigned lds_addr) {
  const unsigned long long ga = (unsigned long long)gbase;
  v4u g0;
  g0[0] = 1u;                                   // count=1 (valid), user mode
  g0[1] = lds_addr;                             // lds_addr [63:32]
  g0[2] = (unsigned)ga;                         // global_addr low
  g0[3] = ((unsigned)(ga >> 32) & 0x01FFFFFFu)  // global_addr[56:32]
          | (2u << 30);                         // type=2 ("image")
  v8i g1;
  g1[0] = (2 << 16)      // data_size = 4B
        | (1 << 20)      // pad_enable
        | (7 << 22)      // pad_interval: 256 DWORDs (= one 1024B row)
        | (3 << 25);     // pad_amount:   4 DWORDs  (= 16B)
  g1[1] = 256 << 16;     // tensor_dim0 = 256 elements (low16 at [63:48])
  g1[2] = (int)(0xCA00u << 16);  // tensor_dim1 = VPAD=51712 (low16 at [95:80])
  g1[3] = 256 << 16;     // tile_dim0 = 256 ([127:112])
  g1[4] = 32;            // tile_dim1 = 32 rows ([143:128]); tile_dim2 = 0
  g1[5] = 256;           // tensor_dim0_stride = 256 elements ([191:160])
  g1[6] = 0;
  g1[7] = 0;
  const v4i z4 = {0, 0, 0, 0};
#if __clang_major__ >= 23
  const v8i z8 = {0, 0, 0, 0, 0, 0, 0, 0};
  __builtin_amdgcn_tensor_load_to_lds(g0, g1, z4, z4, z8, 0);
#else
  __builtin_amdgcn_tensor_load_to_lds(g0, g1, z4, z4, 0);
#endif
}

DEVFN void wait_tensor_le2() {
#if __has_builtin(__builtin_amdgcn_s_wait_tensorcnt)
  __builtin_amdgcn_s_wait_tensorcnt((short)2);
#else
  asm volatile("s_wait_tensorcnt 0x2" ::: "memory");
#endif
}
DEVFN void wait_tensor_0() {
#if __has_builtin(__builtin_amdgcn_s_wait_tensorcnt)
  __builtin_amdgcn_s_wait_tensorcnt((short)0);
#else
  asm volatile("s_wait_tensorcnt 0x0" ::: "memory");
#endif
}
#endif  // USE_TDM

// ---------------------------------------------------------------------------
// Prep: vocab fp32 -> bf16 hi/lo (padded to VPAD rows) + exact fp32 norms.
// One wave per row; 4 waves per 128-thread block.
// ---------------------------------------------------------------------------
__global__ void __launch_bounds__(128) k_prep_vocab(
    const float* __restrict__ allz,
    bf16* __restrict__ whi, bf16* __restrict__ wlo,
    float* __restrict__ norms)
{
  const int wave = threadIdx.x >> 5, lane = threadIdx.x & 31;
  const int row  = blockIdx.x * 4 + wave;
  if (row >= VPAD) return;
  const size_t base = (size_t)row * DD;
  if (row < VV) {
    float acc = 0.0f;
#pragma unroll
    for (int it = 0; it < 16; ++it) {
      const int i = it * 32 + lane;
      const float x = allz[base + i];
      const bf16 h  = f2bf(x);
      const bf16 lo = f2bf(x - bf2f(h));
      whi[base + i] = h;
      wlo[base + i] = lo;
      acc += x * x;
    }
#pragma unroll
    for (int m = 16; m >= 1; m >>= 1) acc += __shfl_xor(acc, m, 32);
    if (lane == 0) norms[row] = acc;
  } else {
    const bf16 z = f2bf(0.0f);
#pragma unroll
    for (int it = 0; it < 16; ++it) {
      const int i = it * 32 + lane;
      whi[base + i] = z;
      wlo[base + i] = z;
    }
    if (lane == 0) norms[row] = 3.0e37f;   // pad rows can never win argmin
  }
}

// queries fp32 -> bf16 hi/lo
__global__ void __launch_bounds__(128) k_prep_q(
    const float* __restrict__ z,
    bf16* __restrict__ qhi, bf16* __restrict__ qlo)
{
  const int wave = threadIdx.x >> 5, lane = threadIdx.x & 31;
  const int row  = blockIdx.x * 4 + wave;
  if (row >= NQ) return;
  const size_t base = (size_t)row * DD;
#pragma unroll
  for (int it = 0; it < 16; ++it) {
    const int i = it * 32 + lane;
    const float x = z[base + i];
    const bf16 h  = f2bf(x);
    qhi[base + i] = h;
    qlo[base + i] = f2bf(x - bf2f(h));
  }
}

// ---------------------------------------------------------------------------
// Main WMMA argmin kernel.
// grid = (NQ/64, NCHUNK), block = 128 threads = 4 waves.
// Wave w owns queries [blockIdx.x*64 + w*16, +16). A fragments (hi+lo, all
// 16 k-steps) stay resident in VGPRs. Vocab tiles (32 rows, hi+lo bf16) are
// DMA'd into a double-buffered LDS region (TDM or async-to-LDS), then each
// wave runs 16 k-steps x 6 WMMA (2 independent N-tiles x {hh, hl, lh}).
// ---------------------------------------------------------------------------
__global__ void __launch_bounds__(128) k_argmin(
    const bf16* __restrict__ whi, const bf16* __restrict__ wlo,
    const float* __restrict__ norms,
    const bf16* __restrict__ qhi, const bf16* __restrict__ qlo,
    float* __restrict__ pval, int* __restrict__ pidx)
{
  extern __shared__ __align__(16) unsigned char smem[];   // SMEM_BYTES

  const int tid  = threadIdx.x;
  const int wave = tid >> 5;
  const int lane = tid & 31;
  const int hh   = lane >> 4;          // half of the wave (0/1)
  const int l16  = lane & 15;
  const int qbase = blockIdx.x * 64 + wave * 16;
  const int qrow  = qbase + l16;
  const int chunkStart = blockIdx.y * ROWS_PER_CHUNK;

  // DMA one tile (32 vocab rows, hi+lo planes) into buffer `buf`.
  auto stage_tile = [&](int r0, int buf) {
    const unsigned lbase = (unsigned)(uintptr_t)(smem + buf * LBUF_BYTES);
#if USE_TDM
    if (wave == 0) {
      tdm_load_plane(whi + (size_t)r0 * DD, lbase);
      tdm_load_plane(wlo + (size_t)r0 * DD, lbase + PLANE_BYTES);
    }
#else
    const unsigned char* ghi = (const unsigned char*)(whi + (size_t)r0 * DD);
    const unsigned char* glo = (const unsigned char*)(wlo + (size_t)r0 * DD);
#pragma unroll
    for (int i = 0; i < 16; ++i) {
      const int idx = i * 128 + tid;        // 0..2047 16B chunks per plane
      const int row = idx >> 6;
      const int c   = idx & 63;
      const unsigned dst = lbase + (unsigned)(row * LDS_ROW_STRIDE + c * 16);
      const size_t  goff = (size_t)row * 1024 + (size_t)c * 16;
      async_cp16(dst,               ghi + goff);   // hi plane
      async_cp16(dst + PLANE_BYTES, glo + goff);   // lo plane
    }
#endif
  };
  auto drain_prev = [&](bool more_inflight) {
#if USE_TDM
    if (wave == 0) { if (more_inflight) wait_tensor_le2(); else wait_tensor_0(); }
#else
    if (more_inflight) wait_async_le32(); else wait_async_0();
#endif
  };

  // kick off DMA for tile 0 before doing anything else
  stage_tile(chunkStart, 0);

  // ---- resident A fragments: 16 k-steps, hi + lo ----------------------
  // 16-bit A 16x32 layout: lane m (m<16) K = 0..7 & 16..23 ; lane m+16
  // K = 8..15 & 24..31 (per k-step of 32).
  BF16x16 ahi[16], alo[16];
  {
    const int kb = hh * 8;
    const size_t rb = (size_t)qrow * DD;
#pragma unroll
    for (int s = 0; s < 16; ++s) {
      const size_t o = rb + (size_t)(s * 32 + kb);
      ahi[s].h[0] = *(const v8bf*)(qhi + o);
      ahi[s].h[1] = *(const v8bf*)(qhi + o + 16);
      alo[s].h[0] = *(const v8bf*)(qlo + o);
      alo[s].h[1] = *(const v8bf*)(qlo + o + 16);
    }
  }

  float mval[8];
  int   midx[8];
#pragma unroll
  for (int r = 0; r < 8; ++r) { mval[r] = 3.4e38f; midx[r] = 0; }

  for (int t = 0; t < TILES_PER_CHUNK; ++t) {
    const int r0  = chunkStart + t * VT;
    const int cur = t & 1;

    // prefetch next tile into the other buffer (its last readers finished at
    // the end-of-compute barrier of iteration t-1), then drain tile t's DMA.
    if (t + 1 < TILES_PER_CHUNK) {
      stage_tile(r0 + VT, cur ^ 1);
      drain_prev(true);
    } else {
      drain_prev(false);
    }
    __syncthreads();          // all DMA for tile t complete, visible to block

    unsigned char* const lhi = smem + cur * LBUF_BYTES;
    unsigned char* const llo = lhi + PLANE_BYTES;

    v8f acc0 = {};   // N-tile rows r0..r0+15
    v8f acc1 = {};   // N-tile rows r0+16..r0+31
#pragma unroll
    for (int s = 0; s < 16; ++s) {
      // 16-bit B 32x16 layout: lane n (n<16) K = 0..7 & 16..23, lane n+16 K = 8..15 & 24..31
      // -> per-lane two 16B pieces at element offsets s*32 + hh*8 and +16.
      const int boff = s * 64 + hh * 16;          // bytes within row
      const int rb0 = l16 * LDS_ROW_STRIDE + boff;
      const int rb1 = (16 + l16) * LDS_ROW_STRIDE + boff;
      BF16x16 bh0, bl0, bh1, bl1;
      bh0.h[0] = *(const v8bf*)(lhi + rb0);
      bh0.h[1] = *(const v8bf*)(lhi + rb0 + 32);
      bl0.h[0] = *(const v8bf*)(llo + rb0);
      bl0.h[1] = *(const v8bf*)(llo + rb0 + 32);
      bh1.h[0] = *(const v8bf*)(lhi + rb1);
      bh1.h[1] = *(const v8bf*)(lhi + rb1 + 32);
      bl1.h[0] = *(const v8bf*)(llo + rb1);
      bl1.h[1] = *(const v8bf*)(llo + rb1 + 32);

      // dot ~= hi*hi + hi*lo + lo*hi, all accumulated in fp32.
      acc0 = __builtin_amdgcn_wmma_f32_16x16x32_bf16(false, ahi[s].v, false, bh0.v, (short)0, acc0, false, false);
      acc1 = __builtin_amdgcn_wmma_f32_16x16x32_bf16(false, ahi[s].v, false, bh1.v, (short)0, acc1, false, false);
      acc0 = __builtin_amdgcn_wmma_f32_16x16x32_bf16(false, ahi[s].v, false, bl0.v, (short)0, acc0, false, false);
      acc1 = __builtin_amdgcn_wmma_f32_16x16x32_bf16(false, ahi[s].v, false, bl1.v, (short)0, acc1, false, false);
      acc0 = __builtin_amdgcn_wmma_f32_16x16x32_bf16(false, alo[s].v, false, bh0.v, (short)0, acc0, false, false);
      acc1 = __builtin_amdgcn_wmma_f32_16x16x32_bf16(false, alo[s].v, false, bh1.v, (short)0, acc1, false, false);
    }

    // C layout: lane = N%16 + 16*(M>=8), vgpr r -> M = r + 8*(lane>=16)
    const int n0 = r0 + l16;
    const int n1 = n0 + 16;
    const float nm0 = norms[n0];
    const float nm1 = norms[n1];
#pragma unroll
    for (int r = 0; r < 8; ++r) {
      const float s0 = nm0 - 2.0f * acc0[r];
      const float s1 = nm1 - 2.0f * acc1[r];
      if (s0 < mval[r]) { mval[r] = s0; midx[r] = n0; }
      if (s1 < mval[r]) { mval[r] = s1; midx[r] = n1; }
    }
    __syncthreads();   // all waves done reading buf[cur]; safe to DMA into it
  }

  // ---- reduce across the 16 lanes of each half (same M, different N) ----
#pragma unroll
  for (int r = 0; r < 8; ++r) {
    float v = mval[r];
    int   ix = midx[r];
#pragma unroll
    for (int m = 8; m >= 1; m >>= 1) {
      const float ov = __shfl_xor(v, m, 32);
      const int   oi = __shfl_xor(ix, m, 32);
      if (ov < v) { v = ov; ix = oi; }
    }
    if (l16 == 0) {
      const int q = qbase + hh * 8 + r;       // half0 -> M=r, half1 -> M=r+8
      pval[(size_t)blockIdx.y * NQ + q] = v;
      pidx[(size_t)blockIdx.y * NQ + q] = ix;
    }
  }
}

// ---------------------------------------------------------------------------
// Finalize: reduce NCHUNK partials per query, gather fp32 row from all_z.
// One wave per query.
// ---------------------------------------------------------------------------
__global__ void __launch_bounds__(128) k_finalize(
    const float* __restrict__ pval, const int* __restrict__ pidx,
    const float* __restrict__ allz, float* __restrict__ out)
{
  const int wave = threadIdx.x >> 5, lane = threadIdx.x & 31;
  const int q = blockIdx.x * 4 + wave;
  if (q >= NQ) return;
  int best = 0;
  if (lane == 0) {
    float bv = 3.4e38f;
    for (int c = 0; c < NCHUNK; ++c) {
      const float v = pval[(size_t)c * NQ + q];
      const int   i = pidx[(size_t)c * NQ + q];
      if (v < bv) { bv = v; best = i; }
    }
  }
  best = __shfl(best, 0, 32);
  const float4* src = (const float4*)(allz + (size_t)best * DD);
  float4* dst = (float4*)(out + (size_t)q * DD);
#pragma unroll
  for (int j = 0; j < 4; ++j) dst[j * 32 + lane] = src[j * 32 + lane];
}

// ---------------------------------------------------------------------------
extern "C" void kernel_launch(void* const* d_in, const int* in_sizes, int n_in,
                              void* d_out, int out_size, void* d_ws, size_t ws_size,
                              hipStream_t stream)
{
  (void)in_sizes; (void)n_in; (void)out_size; (void)ws_size;
  const float* z    = (const float*)d_in[0];   // [B, L, D] fp32
  const float* allz = (const float*)d_in[1];   // [V, D]    fp32
  float* out = (float*)d_out;                  // [B, L, D] fp32

  unsigned char* ws = (unsigned char*)d_ws;
  size_t off = 0;
  bf16*  whi   = (bf16*)(ws + off);  off += (size_t)VPAD * DD * 2;
  bf16*  wlo   = (bf16*)(ws + off);  off += (size_t)VPAD * DD * 2;
  float* norms = (float*)(ws + off); off += (size_t)VPAD * 4;
  bf16*  qhi   = (bf16*)(ws + off);  off += (size_t)NQ * DD * 2;
  bf16*  qlo   = (bf16*)(ws + off);  off += (size_t)NQ * DD * 2;
  float* pval  = (float*)(ws + off); off += (size_t)NCHUNK * NQ * 4;
  int*   pidx  = (int*)(ws + off);   off += (size_t)NCHUNK * NQ * 4;

  k_prep_vocab<<<dim3(VPAD / 4), dim3(128), 0, stream>>>(allz, whi, wlo, norms);
  k_prep_q   <<<dim3(NQ / 4),   dim3(128), 0, stream>>>(z, qhi, qlo);
  k_argmin   <<<dim3(NQ / 64, NCHUNK), dim3(128), SMEM_BYTES, stream>>>(
                 whi, wlo, norms, qhi, qlo, pval, pidx);
  k_finalize <<<dim3(NQ / 4),   dim3(128), 0, stream>>>(pval, pidx, allz, out);
}
